// ProgressiveLuongAttention_25048249270886
// MI455X (gfx1250) — matrix operand order repaired and checked by reading, not compiled, not run on hardware
//
// ProgressiveLuongAttention for MI455X (gfx1250) — flash-attention formulation.
//
// Roofline: ~35 GFLOP total; materializing att ([8,4096,4096] f32) costs >=1GB
// HBM traffic vs ~50MB for flash streaming -> WMMA-bound. Both GEMMs run on
// v_wmma_f32_16x16x32_f16 (f16 operands, f32 accum). This revision:
//  * BLOCK_N=64: 16 WMMAs per wave-iteration, fewer barriers per column
//  * softmax row reductions via DPP16 ROW_XMASK (pure VALU) instead of
//    ds_bpermute (LDS pipe)
//  * V tile staged with GLOBAL_LOAD_ASYNC_TO_LDS_B128 (ASYNCcnt) — builtin
//    signature takes int4* pointers (global AS, LDS AS) per the round-2
//    diagnostic; K tile transposed on the fly into padded LDS (stride 72
//    halves breaks the transpose-store bank-conflict pattern)
//  * output transposed through LDS for coalesced b128 stores

#include <hip/hip_runtime.h>
#include <hip/hip_bf16.h>
#include <math.h>

typedef _Float16 f16;
typedef __attribute__((ext_vector_type(16))) _Float16 v16h;
typedef __attribute__((ext_vector_type(8)))  _Float16 v8h;
typedef __attribute__((ext_vector_type(8)))  float    v8f;
typedef __attribute__((ext_vector_type(4)))  int      v4i;

#define BSZ  8
#define CDIM 64
#define NTOK 4096   // H*W
#define BN   64     // attention columns per iteration
#define LDP  72     // padded LDS row stride in halves (144B = 16B-aligned rows)

#if __has_builtin(__builtin_amdgcn_global_load_async_to_lds_b128)
#define HAVE_ASYNC_LDS 1
typedef __attribute__((address_space(1))) v4i* gp1;   // global-side int4*
typedef __attribute__((address_space(3))) v4i* lp3;   // LDS-side int4*
#else
#define HAVE_ASYNC_LDS 0
#endif

// ---- 16-lane butterfly reductions on the VALU via DPP16 ROW_XMASK ----------
template <int MASK>
__device__ __forceinline__ float xmask16(float v) {
    return __builtin_bit_cast(float,
        __builtin_amdgcn_update_dpp(0, __builtin_bit_cast(int, v),
                                    0x160 | MASK, 0xF, 0xF, true));
}
__device__ __forceinline__ float row16_max(float v) {
    v = fmaxf(v, xmask16<1>(v));
    v = fmaxf(v, xmask16<2>(v));
    v = fmaxf(v, xmask16<4>(v));
    v = fmaxf(v, xmask16<8>(v));
    return v;
}
__device__ __forceinline__ float row16_sum(float v) {
    v += xmask16<1>(v);
    v += xmask16<2>(v);
    v += xmask16<4>(v);
    v += xmask16<8>(v);
    return v;
}

__device__ __forceinline__ v16h cat8(v8h lo, v8h hi) {
    v16h r;
    #pragma unroll
    for (int h = 0; h < 8; ++h) { r[h] = lo[h]; r[8 + h] = hi[h]; }
    return r;
}

// ---------------------------------------------------------------------------
// Kernel 1: q/k/v = W·act + b, f16 output in natural [B][C][N] layout (~2% of
// FLOPs; weights broadcast from LDS, activations/outputs fully coalesced).
// ---------------------------------------------------------------------------
__global__ __launch_bounds__(256)
void qkv_proj_kernel(const float* __restrict__ x, const float* __restrict__ y,
                     const float* __restrict__ Wq, const float* __restrict__ bq,
                     const float* __restrict__ Wk, const float* __restrict__ bk,
                     const float* __restrict__ Wv, const float* __restrict__ bv,
                     f16* __restrict__ qw, f16* __restrict__ kw, f16* __restrict__ vw)
{
    __shared__ __attribute__((aligned(16))) float Ws[CDIM * CDIM];
    __shared__ float bs[CDIM];

    const int tid  = threadIdx.x;
    const int proj = blockIdx.y;       // 0:q 1:k 2:v
    const int b    = blockIdx.z;

    const float* W    = (proj == 0) ? Wq : (proj == 1) ? Wk : Wv;
    const float* bias = (proj == 0) ? bq : (proj == 1) ? bk : bv;
    const float* A    = (proj == 0) ? x  : y;     // q reads x; k,v read y
    f16*         O    = (proj == 0) ? qw : (proj == 1) ? kw : vw;

    for (int s = tid; s < CDIM * CDIM; s += 256) Ws[s] = W[s];
    if (tid < CDIM) bs[tid] = bias[tid];
    __syncthreads();

    const int hw = blockIdx.x * 256 + tid;
    const float* ac = A + (size_t)b * CDIM * NTOK + hw;

    float xr[CDIM];
    #pragma unroll
    for (int ci = 0; ci < CDIM; ++ci) xr[ci] = ac[(size_t)ci * NTOK];

    f16* oc = O + (size_t)b * CDIM * NTOK + hw;
    for (int co = 0; co < CDIM; ++co) {
        float acc = bs[co];
        #pragma unroll
        for (int ci = 0; ci < CDIM; ++ci)
            acc = fmaf(Ws[co * CDIM + ci], xr[ci], acc);
        oc[(size_t)co * NTOK] = (f16)acc;
    }
}

// ---------------------------------------------------------------------------
// Kernel 2: flash attention. Grid (64 row-blocks, 8 batches), 128 threads.
// Wave w owns a 16-row strip; streams 4096 columns in blocks of 64.
// ---------------------------------------------------------------------------
__global__ __launch_bounds__(128)
void flash_attn_kernel(const f16* __restrict__ qw, const f16* __restrict__ kw,
                       const f16* __restrict__ vw, const float* __restrict__ x,
                       const float* __restrict__ gamma, float* __restrict__ out)
{
    __shared__ __attribute__((aligned(64))) f16  Kt[BN][LDP];       // [m][c] transposed
    __shared__ __attribute__((aligned(64))) f16  Vs[CDIM][LDP];     // [c][j]
    __shared__ __attribute__((aligned(64))) f16  Ps[4][16][LDP];    // per-wave P
    __shared__ __attribute__((aligned(16))) float Ot[CDIM][CDIM];   // out transpose

    const int tid = threadIdx.x;
    const int wv  = tid >> 5;        // wave 0..3
    const int ln  = tid & 31;
    const int lh  = ln >> 4;         // half-wave 0/1
    const int lm  = ln & 15;
    const int b   = blockIdx.y;
    const int ib0 = blockIdx.x * 64;           // first att row of this block
    const int i0  = ib0 + wv * 16;             // first att row of this wave

    const f16* qb = qw + (size_t)b * CDIM * NTOK;
    const f16* kb = kw + (size_t)b * CDIM * NTOK;
    const f16* vb = vw + (size_t)b * CDIM * NTOK;

    // ---- preload Q strip as two 16x32 A-operands (K = c, 0..31 / 32..63) ----
    // qf[n,c] is the flat q buffer read as [4096,64] row-major.  A-layout:
    // lane l -> row l%16, halves[h] = K[16*(l/16) + h].
    const f16* qrow = qb + (size_t)(i0 + lm) * 64;
    const v16h qa0 = cat8(*(const v8h*)(qrow +  0 +     16 * lh),
                          *(const v8h*)(qrow +  0 + 8 + 16 * lh));
    const v16h qa1 = cat8(*(const v8h*)(qrow + 32 +     16 * lh),
                          *(const v8h*)(qrow + 32 + 8 + 16 * lh));

    float m_run[8], l_run[8];
    v8f  o[4];
    #pragma unroll
    for (int g = 0; g < 8; ++g) { m_run[g] = -1e30f; l_run[g] = 0.0f; }
    #pragma unroll
    for (int cc = 0; cc < 4; ++cc)
        #pragma unroll
        for (int g = 0; g < 8; ++g) o[cc][g] = 0.0f;

    for (int it = 0; it < NTOK / BN; ++it) {
        const int j0 = it * BN;

        // ---- stage K (transposed) and V blocks into LDS -------------------
        __syncthreads();   // previous iteration's LDS reads are done
        #pragma unroll
        for (int r = 0; r < 4; ++r) {
            const int ch = r * 128 + tid;      // 0..511 chunks of 8 halves
            const int c  = ch >> 3;            // 0..63
            const int of = (ch & 7) * 8;       // m_local base 0..56
            v8h k8 = *(const v8h*)(kb + (size_t)c * NTOK + j0 + of);
            #pragma unroll
            for (int e = 0; e < 8; ++e) Kt[of + e][c] = k8[e];
#if HAVE_ASYNC_LDS
            __builtin_amdgcn_global_load_async_to_lds_b128(
                (gp1)(vb + (size_t)c * NTOK + j0 + of),
                (lp3)&Vs[c][of], 0, 0);
#else
            *(v8h*)&Vs[c][of] = *(const v8h*)(vb + (size_t)c * NTOK + j0 + of);
#endif
        }
#if HAVE_ASYNC_LDS
        asm volatile("s_wait_asynccnt 0x0" ::: "memory");
#endif
        __syncthreads();

        // prefetch next K/V block while we compute on this one
        if (it + 1 < NTOK / BN) {
            const int pj = j0 + BN;
            const int pc = tid >> 1;
            const int ph = (tid & 1) * 32;
            __builtin_prefetch(kb + (size_t)pc * NTOK + pj + ph, 0, 1);
            __builtin_prefetch(vb + (size_t)pc * NTOK + pj + ph, 0, 1);
        }

        // ---- S = Q(16x64) x K(64x64): four 16x16 tiles, K-dim split 2x32 ---
        v8f s[4];
        #pragma unroll
        for (int t = 0; t < 4; ++t)
            #pragma unroll
            for (int g = 0; g < 8; ++g) s[t][g] = 0.0f;

        #pragma unroll
        for (int kk = 0; kk < 2; ++kk) {
            const v16h a = kk ? qa1 : qa0;
            #pragma unroll
            for (int t = 0; t < 4; ++t) {
                // B-layout: lane l -> col l%16, halves[j] = K[16*(l/16)+j]
                const f16* bp = &Kt[t * 16 + lm][kk * 32 + 16 * lh];
                const v16h bb = cat8(*(const v8h*)bp, *(const v8h*)(bp + 8));
                s[t] = __builtin_amdgcn_wmma_f32_16x16x32_f16(false, a, false, bb,
                                                              (short)0, s[t], false, false);
            }
        }

        // ---- online softmax: rows = M = g + 8*lh, cols = lanes ------------
        float corr[8];
        #pragma unroll
        for (int g = 0; g < 8; ++g) {
            float v = fmaxf(fmaxf(s[0][g], s[1][g]), fmaxf(s[2][g], s[3][g]));
            v = row16_max(v);
            const float mn = fmaxf(m_run[g], v);
            corr[g]  = __expf(m_run[g] - mn);
            m_run[g] = mn;
            float r = 0.0f;
            #pragma unroll
            for (int t = 0; t < 4; ++t) {
                s[t][g] = __expf(s[t][g] - mn);   // s becomes P
                r += s[t][g];
            }
            r = row16_sum(r);
            l_run[g] = l_run[g] * corr[g] + r;
        }
        #pragma unroll
        for (int cc = 0; cc < 4; ++cc)
            #pragma unroll
            for (int g = 0; g < 8; ++g) o[cc][g] *= corr[g];

        // ---- P: D-layout -> f16 A-layout via per-wave LDS scratch ---------
        #pragma unroll
        for (int g = 0; g < 8; ++g) {
            const int r = g + 8 * lh;
            #pragma unroll
            for (int t = 0; t < 4; ++t)
                Ps[wv][r][t * 16 + lm] = (f16)s[t][g];
        }
        const v16h pa0 = cat8(*(const v8h*)&Ps[wv][lm][ 0 +     16 * lh],
                              *(const v8h*)&Ps[wv][lm][ 0 + 8 + 16 * lh]);
        const v16h pa1 = cat8(*(const v8h*)&Ps[wv][lm][32 +     16 * lh],
                              *(const v8h*)&Ps[wv][lm][32 + 8 + 16 * lh]);

        // ---- O[i, c] += P(16x64) x V^T(64x16) per 16-channel chunk --------
        #pragma unroll
        for (int jh = 0; jh < 2; ++jh) {
            const v16h pa = jh ? pa1 : pa0;
            #pragma unroll
            for (int cc = 0; cc < 4; ++cc) {
                const f16* bp = &Vs[cc * 16 + lm][jh * 32 + 16 * lh];
                const v16h bv = cat8(*(const v8h*)bp, *(const v8h*)(bp + 8));
                o[cc] = __builtin_amdgcn_wmma_f32_16x16x32_f16(false, pa, false, bv,
                                                               (short)0, o[cc], false, false);
            }
        }
    }

    // ---- finalize: 1/l, transpose through LDS, coalesced gamma*O + x ------
    float inv[8];
    #pragma unroll
    for (int g = 0; g < 8; ++g) inv[g] = 1.0f / l_run[g];

    #pragma unroll
    for (int cc = 0; cc < 4; ++cc)
        #pragma unroll
        for (int g = 0; g < 8; ++g)
            Ot[cc * 16 + lm][wv * 16 + g + 8 * lh] = o[cc][g] * inv[g];
    __syncthreads();

    const float gm = gamma[0];
    const int   c  = tid >> 1;                 // 0..63
    const int   hh = (tid & 1) * 32;           // half of the 64-row block
    const float* xrow = x   + ((size_t)b * CDIM + c) * NTOK + ib0 + hh;
    float*       orow = out + ((size_t)b * CDIM + c) * NTOK + ib0 + hh;
    #pragma unroll
    for (int e = 0; e < 32; ++e)
        orow[e] = gm * Ot[c][hh + e] + xrow[e];
}

// ---------------------------------------------------------------------------
extern "C" void kernel_launch(void* const* d_in, const int* in_sizes, int n_in,
                              void* d_out, int out_size, void* d_ws, size_t ws_size,
                              hipStream_t stream) {
    const float* x     = (const float*)d_in[0];
    const float* y     = (const float*)d_in[1];
    const float* Wq    = (const float*)d_in[2];
    const float* bq    = (const float*)d_in[3];
    const float* Wk    = (const float*)d_in[4];
    const float* bk    = (const float*)d_in[5];
    const float* Wv    = (const float*)d_in[6];
    const float* bv    = (const float*)d_in[7];
    const float* gamma = (const float*)d_in[8];
    float*       out   = (float*)d_out;

    // workspace: q,k,v in f16, [B][C][N] each = 4 MiB -> 12.6 MB total
    f16* qw = (f16*)d_ws;
    f16* kw = qw + (size_t)BSZ * CDIM * NTOK;
    f16* vw = kw + (size_t)BSZ * CDIM * NTOK;

    dim3 g1(NTOK / 256, 3, BSZ);       // (col-chunk, proj, batch)
    qkv_proj_kernel<<<g1, 256, 0, stream>>>(x, y, Wq, bq, Wk, bk, Wv, bv,
                                            qw, kw, vw);

    dim3 g2(NTOK / 64, BSZ);           // (row-block of 64, batch)
    flash_attn_kernel<<<g2, 128, 0, stream>>>(qw, kw, vw, x, gamma, out);
}
